// EBEPModelWrapper_15496242004064
// MI455X (gfx1250) — matrix operand order repaired and checked by reading, not compiled, not run on hardware
//
#include <hip/hip_runtime.h>
#include <hip/hip_bf16.h>

// ---------------------------------------------------------------------------
// MI455X / gfx1250 EP relaxation.
// - All GEMM work via v_wmma_f32_16x16x32_bf16 (wave32, batch-as-M tiles).
// - States / drives stored bf16 NHWC: ~110MB working set -> L2-resident
//   (192MB), pure b128 loads for A fragments (no f32->bf16 cvt on loads).
// - FC-forward stages its B panel through LDS with the Tensor Data Mover
//   (tensor_load_to_lds + s_wait_tensorcnt), double-buffered.
// ---------------------------------------------------------------------------

typedef __bf16 bf16_t;
typedef __attribute__((ext_vector_type(16))) __bf16 v16bf;
typedef __attribute__((ext_vector_type(8)))  float  v8f;
typedef unsigned int v4u __attribute__((ext_vector_type(4)));
typedef int          v8i __attribute__((ext_vector_type(8)));
typedef int          v4i __attribute__((ext_vector_type(4)));

#define DEV __device__ __forceinline__

#if __has_builtin(__builtin_amdgcn_tensor_load_to_lds) && \
    __has_builtin(__builtin_amdgcn_s_wait_tensorcnt)
#define USE_TDM 1
#else
#define USE_TDM 0
#endif

constexpr int NB   = 4096;
constexpr int S1S  = 14 * 14 * 32;   // 6272 elems / sample
constexpr int S2S  = 7 * 7 * 64;     // 3136 elems / sample (s2, s3)
constexpr int S4S  = 256;
constexpr int KFC  = 3136;
constexpr int TSTEPS = 50;

// packed-weight element counts (bf16): layout [tap|kc][nt][lane(32)][16]
constexpr int PK_C2F = 9 * 1 * 4 * 512;
constexpr int PK_C2T = 9 * 2 * 2 * 512;
constexpr int PK_C3F = 9 * 2 * 4 * 512;
constexpr int PK_C3T = 9 * 2 * 4 * 512;
constexpr int PK_FCF = 98 * 16 * 512;
constexpr int PK_FCB = 8 * 196 * 512;

DEV v8f v8zero() { v8f v = {0.f,0.f,0.f,0.f,0.f,0.f,0.f,0.f}; return v; }

// A-fragment (16x32 bf16 MxK): lane l holds row M=l&15; lanes 0-15 take K
// {0..7}∪{16..23}, lanes 16-31 take {8..15}∪{24..31}. Pointer is pre-offset
// by (lane>=16 ? 8 : 0) channels -> two contiguous 16B runs (b128 loads).
DEV v16bf a_frag_load(const bf16_t* __restrict__ p) {
  v16bf a;
#pragma unroll
  for (int j = 0; j < 8; ++j) {
    a[j]     = p[j];
    a[j + 8] = p[j + 16];
  }
  return a;
}

DEV v8f wmma_bf16(v16bf a, v16bf b, v8f c) {
  return __builtin_amdgcn_wmma_f32_16x16x32_bf16(false, a, false, b, (short)0, c,
                                                 false, false);
}

// ---- conv forward accumulate ----------------------------------------------
template<int CIN, int COUT, int HIN, int WIN, int STRIDE>
DEV void conv_fwd_accum(const bf16_t* __restrict__ sin, const bf16_t* __restrict__ wpk,
                        int b0, int oy, int ox, v8f* acc, int lane)
{
  constexpr int KC = CIN / 32, NT = COUT / 16;
  const int row = lane & 15, half = lane >> 4;
  const bf16_t* sb = sin + (long)(b0 + row) * (HIN * WIN * CIN) + half * 8;
#pragma unroll
  for (int ky = 0; ky < 3; ++ky) {
    int iy = STRIDE * oy - 1 + ky;
    if (iy < 0 || iy >= HIN) continue;
#pragma unroll
    for (int kx = 0; kx < 3; ++kx) {
      int ix = STRIDE * ox - 1 + kx;
      if (ix < 0 || ix >= WIN) continue;
      int tap = ky * 3 + kx;
#pragma unroll
      for (int kc = 0; kc < KC; ++kc) {
        v16bf a = a_frag_load(sb + (iy * WIN + ix) * CIN + kc * 32);
        const bf16_t* bp = wpk + (size_t)((tap * KC + kc) * NT) * 512 + lane * 16;
#pragma unroll
        for (int nt = 0; nt < NT; ++nt) {
          v16bf b = *reinterpret_cast<const v16bf*>(bp + nt * 512);
          acc[nt] = wmma_bf16(a, b, acc[nt]);
        }
      }
    }
  }
}

// ---- transposed-conv accumulate -------------------------------------------
template<int NOUT, int KSRC, int HS, int WS, int STRIDE>
DEV void conv_tr_accum(const bf16_t* __restrict__ snx, const bf16_t* __restrict__ wpk,
                       int b0, int iy, int ix, v8f* acc, int lane)
{
  constexpr int KC = KSRC / 32, NT = NOUT / 16;
  const int row = lane & 15, half = lane >> 4;
  const bf16_t* sb = snx + (long)(b0 + row) * (HS * WS * KSRC) + half * 8;
#pragma unroll
  for (int ky = 0; ky < 3; ++ky) {
    int ty = iy + 1 - ky;
    if (STRIDE == 2 && (ty & 1)) continue;
    int oy = (STRIDE == 2) ? (ty >> 1) : ty;
    if (oy < 0 || oy >= HS) continue;
#pragma unroll
    for (int kx = 0; kx < 3; ++kx) {
      int tx = ix + 1 - kx;
      if (STRIDE == 2 && (tx & 1)) continue;
      int ox = (STRIDE == 2) ? (tx >> 1) : tx;
      if (ox < 0 || ox >= WS) continue;
      int tap = ky * 3 + kx;
#pragma unroll
      for (int kc = 0; kc < KC; ++kc) {
        v16bf a = a_frag_load(sb + (oy * WS + ox) * KSRC + kc * 32);
        const bf16_t* bp = wpk + (size_t)((tap * KC + kc) * NT) * 512 + lane * 16;
#pragma unroll
        for (int nt = 0; nt < NT; ++nt) {
          v16bf b = *reinterpret_cast<const v16bf*>(bp + nt * 512);
          acc[nt] = wmma_bf16(a, b, acc[nt]);
        }
      }
    }
  }
}

// =============================== weight packing =============================
template<int CIN, int COUT>
__global__ void k_pack_conv_fwd(const float* __restrict__ w, bf16_t* __restrict__ pk) {
  int idx = blockIdx.x * 256 + threadIdx.x;
  constexpr int NT = COUT / 16, KC = CIN / 32;
  if (idx >= 9 * KC * NT * 512) return;
  int j = idx & 15, lane = (idx >> 4) & 31, rest = idx >> 9;
  int nt = rest % NT; rest /= NT;
  int kc = rest % KC; int tap = rest / KC;
  int n = nt * 16 + (lane & 15);
  int k = kc * 32 + ((lane >> 4) << 3) + (j & 7) + ((j >> 3) << 4);
  pk[idx] = (bf16_t)w[(n * CIN + k) * 9 + tap];
}

template<int CIN, int COUT>
__global__ void k_pack_conv_tr(const float* __restrict__ w, bf16_t* __restrict__ pk) {
  int idx = blockIdx.x * 256 + threadIdx.x;
  constexpr int NT = CIN / 16, KC = COUT / 32;
  if (idx >= 9 * KC * NT * 512) return;
  int j = idx & 15, lane = (idx >> 4) & 31, rest = idx >> 9;
  int nt = rest % NT; rest /= NT;
  int kc = rest % KC; int tap = rest / KC;
  int n = nt * 16 + (lane & 15);
  int k = kc * 32 + ((lane >> 4) << 3) + (j & 7) + ((j >> 3) << 4);
  pk[idx] = (bf16_t)w[(k * CIN + n) * 9 + tap];
}

__global__ void k_pack_fc_fwd(const float* __restrict__ w4, bf16_t* __restrict__ pk) {
  int idx = blockIdx.x * 256 + threadIdx.x;
  if (idx >= PK_FCF) return;
  int j = idx & 15, lane = (idx >> 4) & 31, rest = idx >> 9;
  int nt = rest & 15, kc = rest >> 4;
  int n = nt * 16 + (lane & 15);
  int k = kc * 32 + ((lane >> 4) << 3) + (j & 7) + ((j >> 3) << 4);
  int col = (k & 63) * 49 + (k >> 6);
  pk[idx] = (bf16_t)w4[(long)n * KFC + col];
}

__global__ void k_pack_fc_bwd(const float* __restrict__ w4, bf16_t* __restrict__ pk) {
  int idx = blockIdx.x * 256 + threadIdx.x;
  if (idx >= PK_FCB) return;
  int j = idx & 15, lane = (idx >> 4) & 31, rest = idx >> 9;
  int nt = rest % 196, kc = rest / 196;
  int n = nt * 16 + (lane & 15);
  int k = kc * 32 + ((lane >> 4) << 3) + (j & 7) + ((j >> 3) << 4);
  int col = (n & 63) * 49 + (n >> 6);
  pk[idx] = (bf16_t)w4[(long)k * KFC + col];
}

// ======================== constant drive: conv1(x)+b1 =======================
__global__ void k_conv1(const float* __restrict__ x, const float* __restrict__ w1,
                        const float* __restrict__ b1, bf16_t* __restrict__ d1c) {
  int t = blockIdx.x * 256 + threadIdx.x;
  if (t >= NB * 196) return;
  int pix = t % 196, b = t / 196;
  int oy = pix / 14, ox = pix % 14;
  float xv[9];
#pragma unroll
  for (int ky = 0; ky < 3; ++ky)
#pragma unroll
    for (int kx = 0; kx < 3; ++kx) {
      int iy = 2 * oy - 1 + ky, ix = 2 * ox - 1 + kx;
      xv[ky * 3 + kx] = (iy >= 0 && iy < 28 && ix >= 0 && ix < 28)
                            ? x[(long)b * 784 + iy * 28 + ix] : 0.f;
    }
  bf16_t* o = d1c + (long)b * S1S + pix * 32;
  for (int co = 0; co < 32; ++co) {
    float s = b1[co];
#pragma unroll
    for (int tap = 0; tap < 9; ++tap) s += w1[co * 9 + tap] * xv[tap];
    o[co] = (bf16_t)s;
  }
}

// ============================= relaxation kernels ===========================
__global__ void k_step_s1(const bf16_t* __restrict__ s2c, const bf16_t* __restrict__ s1c,
                          const bf16_t* __restrict__ d1c, const bf16_t* __restrict__ pk2t,
                          bf16_t* __restrict__ s1n) {
  int lane = threadIdx.x;
  int pix = blockIdx.x, b0 = blockIdx.y * 16;
  int iy = pix / 14, ix = pix % 14;
  v8f acc[2] = {v8zero(), v8zero()};
  conv_tr_accum<32, 64, 7, 7, 2>(s2c, pk2t, b0, iy, ix, acc, lane);
#pragma unroll
  for (int nt = 0; nt < 2; ++nt) {
    int n = nt * 16 + (lane & 15);
#pragma unroll
    for (int r = 0; r < 8; ++r) {
      int m = (lane < 16) ? r : r + 8;
      long i = (long)(b0 + m) * S1S + pix * 32 + n;
      s1n[i] = (bf16_t)(0.5f * (float)s1c[i] + 0.5f * (acc[nt][r] + (float)d1c[i]));
    }
  }
}

__global__ void k_step_s2(const bf16_t* __restrict__ s1c, const bf16_t* __restrict__ s3c,
                          const bf16_t* __restrict__ s2c, const bf16_t* __restrict__ pk2f,
                          const bf16_t* __restrict__ pk3t, const float* __restrict__ b2,
                          bf16_t* __restrict__ s2n) {
  int lane = threadIdx.x;
  int pix = blockIdx.x, b0 = blockIdx.y * 16;
  int oy = pix / 7, ox = pix % 7;
  v8f acc[4] = {v8zero(), v8zero(), v8zero(), v8zero()};
  conv_fwd_accum<32, 64, 14, 14, 2>(s1c, pk2f, b0, oy, ox, acc, lane);
  conv_tr_accum <64, 64, 7, 7, 1>(s3c, pk3t, b0, oy, ox, acc, lane);
#pragma unroll
  for (int nt = 0; nt < 4; ++nt) {
    int n = nt * 16 + (lane & 15);
    float bias = b2[n];
#pragma unroll
    for (int r = 0; r < 8; ++r) {
      int m = (lane < 16) ? r : r + 8;
      long i = (long)(b0 + m) * S2S + pix * 64 + n;
      s2n[i] = (bf16_t)(0.5f * (float)s2c[i] + 0.5f * (acc[nt][r] + bias));
    }
  }
}

__global__ void k_fc_bwd(const bf16_t* __restrict__ s4c, const bf16_t* __restrict__ pkfb,
                         bf16_t* __restrict__ g3) {
  int lane = threadIdx.x;
  int ntg = blockIdx.x, b0 = blockIdx.y * 16;
  const int row = lane & 15, half = lane >> 4;
  v8f acc[7];
#pragma unroll
  for (int t = 0; t < 7; ++t) acc[t] = v8zero();
  const bf16_t* ab = s4c + (long)(b0 + row) * S4S + half * 8;
#pragma unroll
  for (int kc = 0; kc < 8; ++kc) {
    v16bf a = a_frag_load(ab + kc * 32);
#pragma unroll
    for (int t = 0; t < 7; ++t) {
      int nt = ntg * 7 + t;
      const bf16_t* bp = pkfb + (size_t)(kc * 196 + nt) * 512 + lane * 16;
      v16bf b = *reinterpret_cast<const v16bf*>(bp);
      acc[t] = wmma_bf16(a, b, acc[t]);
    }
  }
#pragma unroll
  for (int t = 0; t < 7; ++t) {
    int n = (ntg * 7 + t) * 16 + (lane & 15);
#pragma unroll
    for (int r = 0; r < 8; ++r) {
      int m = (lane < 16) ? r : r + 8;
      g3[(long)(b0 + m) * KFC + n] = (bf16_t)acc[t][r];
    }
  }
}

__global__ void k_step_s3(const bf16_t* __restrict__ s2c, const bf16_t* __restrict__ s3c,
                          const bf16_t* __restrict__ pk3f, const float* __restrict__ b3,
                          const bf16_t* __restrict__ g3, bf16_t* __restrict__ s3n) {
  int lane = threadIdx.x;
  int pix = blockIdx.x, b0 = blockIdx.y * 16;
  int oy = pix / 7, ox = pix % 7;
  v8f acc[4] = {v8zero(), v8zero(), v8zero(), v8zero()};
  conv_fwd_accum<64, 64, 7, 7, 1>(s2c, pk3f, b0, oy, ox, acc, lane);
#pragma unroll
  for (int nt = 0; nt < 4; ++nt) {
    int n = nt * 16 + (lane & 15);
    float bias = b3[n];
#pragma unroll
    for (int r = 0; r < 8; ++r) {
      int m = (lane < 16) ? r : r + 8;
      long i = (long)(b0 + m) * S2S + pix * 64 + n;
      s3n[i] = (bf16_t)(0.5f * (float)s3c[i] +
                        0.5f * (acc[nt][r] + bias + (float)g3[i]));
    }
  }
}

// s4' = 0.5*s4 + 0.5*(s3 @ w4^T + b4); B panel staged through LDS via TDM.
__global__ void k_fc_fwd(const bf16_t* __restrict__ s3c, const bf16_t* __restrict__ s4c,
                         const bf16_t* __restrict__ pkff, const float* __restrict__ b4,
                         bf16_t* __restrict__ s4n) {
  int lane = threadIdx.x;
  int b0 = blockIdx.x * 16, ntg = blockIdx.y;   // ntg in {0,1}: 8 n-tiles each
  const int row = lane & 15, half = lane >> 4;
  v8f acc[8];
#pragma unroll
  for (int t = 0; t < 8; ++t) acc[t] = v8zero();
  const bf16_t* ab = s3c + (long)(b0 + row) * KFC + half * 8;

#if USE_TDM
  __shared__ bf16_t lb[2][8 * 512];             // 2 x 8KB double buffer
  typedef __attribute__((address_space(3))) char* lds_ptr_t;
  unsigned lds_base =
      (unsigned)(unsigned long)(lds_ptr_t)(void*)&lb[0][0];
  unsigned long gbase =
      (unsigned long)(const void*)(pkff + (size_t)(ntg * 8) * 512);
  auto tdm_issue = [&](int kc, int pb) {
    unsigned long ga = gbase + (unsigned long)kc * (16 * 512 * 2); // bytes
    // D# group0: count=1 | lds_addr | global_addr | type=2 ("image")
    v4u g0 = {1u, lds_base + (unsigned)pb * 8192u,
              (unsigned)ga, (unsigned)(ga >> 32) | (2u << 30)};
    // D# group1: data_size=2B; tensor_dim0=4096; tile_dim0=4096 (1-D tile)
    v8i g1 = {(int)(1u << 16), (int)(4096u << 16), 0, (int)(4096u << 16),
              0, 0, 0, 0};
    v4i z4 = {0, 0, 0, 0};
    v8i z8 = {0, 0, 0, 0, 0, 0, 0, 0};
    __builtin_amdgcn_tensor_load_to_lds(g0, g1, z4, z4, z8, 0);
  };
  tdm_issue(0, 0);
#endif

  for (int kc = 0; kc < 98; ++kc) {
    __builtin_prefetch(ab + (kc + 2) * 32, 0, 1);     // global_prefetch_b8
    v16bf a = a_frag_load(ab + kc * 32);
#if USE_TDM
    int pb = kc & 1;
    if (kc + 1 < 98) {
      asm volatile("s_wait_dscnt 0" ::: "memory");    // WAR vs pending ds reads
      tdm_issue(kc + 1, pb ^ 1);
      __builtin_amdgcn_s_wait_tensorcnt((short)1);    // current buffer ready
    } else {
      __builtin_amdgcn_s_wait_tensorcnt((short)0);
    }
    const bf16_t* bbase = &lb[pb][0];
#pragma unroll
    for (int t = 0; t < 8; ++t) {
      v16bf b = *reinterpret_cast<const v16bf*>(bbase + (t * 32 + lane) * 16);
      acc[t] = wmma_bf16(a, b, acc[t]);
    }
#else
#pragma unroll
    for (int t = 0; t < 8; ++t) {
      int nt = ntg * 8 + t;
      const bf16_t* bp = pkff + (size_t)(kc * 16 + nt) * 512 + lane * 16;
      v16bf b = *reinterpret_cast<const v16bf*>(bp);
      acc[t] = wmma_bf16(a, b, acc[t]);
    }
#endif
  }

#pragma unroll
  for (int t = 0; t < 8; ++t) {
    int n = (ntg * 8 + t) * 16 + (lane & 15);
    float bias = b4[n];
#pragma unroll
    for (int r = 0; r < 8; ++r) {
      int m = (lane < 16) ? r : r + 8;
      long i = (long)(b0 + m) * S4S + n;
      s4n[i] = (bf16_t)(0.5f * (float)s4c[i] + 0.5f * (acc[t][r] + bias));
    }
  }
}

// ========================= final per-sample energy ==========================
__global__ void k_energy(const float* __restrict__ x,  const float* __restrict__ vb,
                         const bf16_t* __restrict__ s1, const bf16_t* __restrict__ s2,
                         const bf16_t* __restrict__ s3, const bf16_t* __restrict__ s4,
                         const bf16_t* __restrict__ d1c,
                         const float* __restrict__ w2, const float* __restrict__ b2,
                         const float* __restrict__ w3, const float* __restrict__ b3,
                         const float* __restrict__ w4, const float* __restrict__ b4,
                         float* __restrict__ out) {
  int b = blockIdx.x, tid = threadIdx.x;
  float part = 0.f;
  const float* xb = x + (long)b * 784;
  for (int i = tid; i < 784; i += 256) { float v = xb[i]; part += v * (0.5f * v - vb[i]); }
  const bf16_t* s1b = s1 + (long)b * S1S;  const bf16_t* d1b = d1c + (long)b * S1S;
  for (int i = tid; i < S1S; i += 256) {
    float v = (float)s1b[i]; part += 0.5f * v * v - v * (float)d1b[i];
  }
  const bf16_t* s2b = s2 + (long)b * S2S;  const bf16_t* s3b = s3 + (long)b * S2S;
  const bf16_t* s4b = s4 + (long)b * S4S;
  for (int o = tid; o < S2S; o += 256) {
    int pix = o >> 6, co = o & 63, oy = pix / 7, ox = pix % 7;
    float c = b2[co];
    for (int ky = 0; ky < 3; ++ky) {
      int iy = 2 * oy - 1 + ky; if (iy < 0 || iy >= 14) continue;
      for (int kx = 0; kx < 3; ++kx) {
        int ix = 2 * ox - 1 + kx; if (ix < 0 || ix >= 14) continue;
        const bf16_t* sp = s1b + (iy * 14 + ix) * 32;
        const float* wp = w2 + co * 288 + ky * 3 + kx;
        for (int ci = 0; ci < 32; ++ci) c += wp[ci * 9] * (float)sp[ci];
      }
    }
    float v = (float)s2b[o]; part += 0.5f * v * v - v * c;
  }
  for (int o = tid; o < S2S; o += 256) {
    int pix = o >> 6, co = o & 63, oy = pix / 7, ox = pix % 7;
    float c = b3[co];
    for (int ky = 0; ky < 3; ++ky) {
      int iy = oy - 1 + ky; if (iy < 0 || iy >= 7) continue;
      for (int kx = 0; kx < 3; ++kx) {
        int ix = ox - 1 + kx; if (ix < 0 || ix >= 7) continue;
        const bf16_t* sp = s2b + (iy * 7 + ix) * 64;
        const float* wp = w3 + co * 576 + ky * 3 + kx;
        for (int ci = 0; ci < 64; ++ci) c += wp[ci * 9] * (float)sp[ci];
      }
    }
    float v = (float)s3b[o]; part += 0.5f * v * v - v * c;
  }
  for (int o = tid; o < S4S; o += 256) {
    float c = b4[o];
    const float* wr = w4 + (long)o * KFC;
    for (int p = 0; p < 49; ++p)
      for (int ci = 0; ci < 64; ++ci) c += wr[ci * 49 + p] * (float)s3b[p * 64 + ci];
    float v = (float)s4b[o]; part += 0.5f * v * v - v * c;
  }
  __shared__ float red[256];
  red[tid] = part; __syncthreads();
  for (int s = 128; s > 0; s >>= 1) { if (tid < s) red[tid] += red[tid + s]; __syncthreads(); }
  if (tid == 0) out[b] = red[0];
}

// ================================= host =====================================
extern "C" void kernel_launch(void* const* d_in, const int* in_sizes, int n_in,
                              void* d_out, int out_size, void* d_ws, size_t ws_size,
                              hipStream_t stream) {
  const float* x  = (const float*)d_in[0];
  const float* vb = (const float*)d_in[1];
  const float* w1 = (const float*)d_in[2];
  const float* b1 = (const float*)d_in[3];
  const float* w2 = (const float*)d_in[4];
  const float* b2 = (const float*)d_in[5];
  const float* w3 = (const float*)d_in[6];
  const float* b3 = (const float*)d_in[7];
  const float* w4 = (const float*)d_in[8];
  const float* b4 = (const float*)d_in[9];
  float* out = (float*)d_out;

  const size_t S1B = (size_t)NB * S1S * 2;   // bf16 storage
  const size_t S2B = (size_t)NB * S2S * 2;
  const size_t S4B = (size_t)NB * S4S * 2;

  size_t off = 0;
  auto carve = [&](size_t bytes) {
    void* p = (char*)d_ws + off;
    off += (bytes + 255) & ~(size_t)255;
    return p;
  };
  bf16_t* s1[2] = {(bf16_t*)carve(S1B), (bf16_t*)carve(S1B)};
  bf16_t* s2[2] = {(bf16_t*)carve(S2B), (bf16_t*)carve(S2B)};
  bf16_t* s3[2] = {(bf16_t*)carve(S2B), (bf16_t*)carve(S2B)};
  bf16_t* s4[2] = {(bf16_t*)carve(S4B), (bf16_t*)carve(S4B)};
  bf16_t* d1c = (bf16_t*)carve(S1B);
  bf16_t* g3  = (bf16_t*)carve(S2B);
  bf16_t* pk2f = (bf16_t*)carve(PK_C2F * 2);
  bf16_t* pk2t = (bf16_t*)carve(PK_C2T * 2);
  bf16_t* pk3f = (bf16_t*)carve(PK_C3F * 2);
  bf16_t* pk3t = (bf16_t*)carve(PK_C3T * 2);
  bf16_t* pkff = (bf16_t*)carve(PK_FCF * 2);
  bf16_t* pkfb = (bf16_t*)carve(PK_FCB * 2);

  k_pack_conv_fwd<32, 64><<<(PK_C2F + 255) / 256, 256, 0, stream>>>(w2, pk2f);
  k_pack_conv_tr <32, 64><<<(PK_C2T + 255) / 256, 256, 0, stream>>>(w2, pk2t);
  k_pack_conv_fwd<64, 64><<<(PK_C3F + 255) / 256, 256, 0, stream>>>(w3, pk3f);
  k_pack_conv_tr <64, 64><<<(PK_C3T + 255) / 256, 256, 0, stream>>>(w3, pk3t);
  k_pack_fc_fwd<<<(PK_FCF + 255) / 256, 256, 0, stream>>>(w4, pkff);
  k_pack_fc_bwd<<<(PK_FCB + 255) / 256, 256, 0, stream>>>(w4, pkfb);

  (void)hipMemsetAsync(s1[0], 0, S1B, stream);
  (void)hipMemsetAsync(s2[0], 0, S2B, stream);
  (void)hipMemsetAsync(s3[0], 0, S2B, stream);
  (void)hipMemsetAsync(s4[0], 0, S4B, stream);
  k_conv1<<<(NB * 196 + 255) / 256, 256, 0, stream>>>(x, w1, b1, d1c);

  int cur = 0;
  for (int t = 0; t < TSTEPS; ++t) {
    int nxt = cur ^ 1;
    k_step_s1<<<dim3(196, NB / 16), 32, 0, stream>>>(s2[cur], s1[cur], d1c, pk2t, s1[nxt]);
    k_step_s2<<<dim3(49,  NB / 16), 32, 0, stream>>>(s1[cur], s3[cur], s2[cur], pk2f, pk3t, b2, s2[nxt]);
    k_fc_bwd <<<dim3(28,  NB / 16), 32, 0, stream>>>(s4[cur], pkfb, g3);
    k_step_s3<<<dim3(49,  NB / 16), 32, 0, stream>>>(s2[cur], s3[cur], pk3f, b3, g3, s3[nxt]);
    k_fc_fwd <<<dim3(NB / 16, 2),   32, 0, stream>>>(s3[cur], s4[cur], pkff, b4, s4[nxt]);
    cur = nxt;
  }
  k_energy<<<NB, 256, 0, stream>>>(x, vb, s1[cur], s2[cur], s3[cur], s4[cur], d1c,
                                   w2, b2, w3, b3, w4, b4, out);
}
